// PhiMoEDecoderLayer_7516192768999
// MI455X (gfx1250) — matrix operand order repaired and checked by reading, not compiled, not run on hardware
//
#include <hip/hip_runtime.h>

// ---------------------------------------------------------------- constants
constexpr int E = 8;
constexpr int H = 2048;
constexpr int I = 4096;
constexpr int T = 2048;

constexpr int LDA = H + 8;   // bf16 elems per staged-x row (pad kills bank conflicts)
constexpr int LDI = I + 8;   // bf16 elems per inter row
constexpr size_t SMEM_BYTES = (size_t)16 * LDA * 2 + (size_t)16 * LDI * 2; // ~197 KB

// ---------------------------------------------------------------- types
typedef __bf16 bf16_t;
typedef bf16_t v16bf __attribute__((ext_vector_type(16)));
typedef bf16_t v8bf  __attribute__((ext_vector_type(8)));
typedef float  v8f   __attribute__((ext_vector_type(8)));
typedef float  v4f   __attribute__((ext_vector_type(4)));

__device__ inline v8f zero8() {
    v8f z;
#pragma unroll
    for (int i = 0; i < 8; ++i) z[i] = 0.f;
    return z;
}

// Fast reciprocal: single v_rcp_f32 instead of IEEE div chain.
__device__ inline float fast_rcp(float x) { return __builtin_amdgcn_rcpf(x); }

// A fragment (16x32 bf16, ISA layout): lane holds row m=lane%16,
// k = {c0..c0+7} U {16+c0..16+c0+7}, c0 = (lane>=16)*8.  Two ds_load_b128.
__device__ inline v16bf load_a_frag(const bf16_t* __restrict__ row, int koff) {
    v8bf lo = *(const v8bf*)(row + koff);
    v8bf hi = *(const v8bf*)(row + koff + 16);
    return __builtin_shufflevector(lo, hi, 0, 1, 2, 3, 4, 5, 6, 7,
                                           8, 9, 10, 11, 12, 13, 14, 15);
}

// B fragment (32x16 bf16, ISA layout): lane holds col n=lane%16,
// k = {k0..k0+7} U {16+k0..16+k0+7} with k0 = (lane>=16)*8.  Source is fp32 -> convert to bf16.
__device__ inline v16bf load_b_frag(const float* __restrict__ p) {
    v4f f0 = *(const v4f*)(p);
    v4f f1 = *(const v4f*)(p + 4);
    v4f f2 = *(const v4f*)(p + 16);
    v4f f3 = *(const v4f*)(p + 20);
    v16bf b;
#pragma unroll
    for (int i = 0; i < 4; ++i) {
        b[i]      = (bf16_t)f0[i];
        b[4 + i]  = (bf16_t)f1[i];
        b[8 + i]  = (bf16_t)f2[i];
        b[12 + i] = (bf16_t)f3[i];
    }
    return b;
}

__device__ inline v8f wmma_bf16(v16bf a, v16bf b, v8f c) {
    return __builtin_amdgcn_wmma_f32_16x16x32_bf16(
        /*neg_a=*/false, a, /*neg_b=*/false, b,
        /*c_mod=*/(short)0, c, /*reuse_a=*/false, /*reuse_b=*/false);
}

// Same-A second WMMA of a back-to-back identical pair: set the ISA's
// operand-reuse hint (OPSEL.RA) so the matrix pipe re-uses the cached A operand.
__device__ inline v8f wmma_bf16_reuseA(v16bf a, v16bf b, v8f c) {
    return __builtin_amdgcn_wmma_f32_16x16x32_bf16(
        false, a, false, b, (short)0, c, /*reuse_a=*/true, /*reuse_b=*/false);
}

// ---------------------------------------------------------------- kernel 0: zero out + counters
__global__ __launch_bounds__(256) void moe_zero_kernel(float* __restrict__ out, int n,
                                                       int* __restrict__ counts) {
    int stride = gridDim.x * blockDim.x;
    for (int i = blockIdx.x * blockDim.x + threadIdx.x; i < n; i += stride) out[i] = 0.f;
    if (blockIdx.x == 0 && threadIdx.x < E) counts[threadIdx.x] = 0;
}

// ---------------------------------------------------------------- kernel 1: router + sparsemixer
__global__ __launch_bounds__(256) void moe_router_kernel(const float* __restrict__ x,
                                                         const float* __restrict__ gw,
                                                         int* __restrict__ counts,
                                                         int* __restrict__ toklist,
                                                         float* __restrict__ gatec) {
    int t = blockIdx.x * 256 + threadIdx.x;
    if (t >= T) return;

    float s[E];
#pragma unroll
    for (int e = 0; e < E; ++e) s[e] = 0.f;

    const float* xr = x + (size_t)t * H;
    for (int h = 0; h < H; h += 4) {
        v4f xv = *(const v4f*)(xr + h);
#pragma unroll
        for (int e = 0; e < E; ++e) {
            v4f gv = *(const v4f*)(gw + (size_t)e * H + h);  // uniform -> scalar loads
            s[e] += xv[0] * gv[0] + xv[1] * gv[1] + xv[2] * gv[2] + xv[3] * gv[3];
        }
    }

    const float two_eps = 2.f * 0.01f;

    // ---- top-1 + masked softmax gate
    int i1 = 0; float m1 = s[0];
#pragma unroll
    for (int e = 1; e < E; ++e) if (s[e] > m1) { m1 = s[e]; i1 = e; }
    float sum1 = 0.f;
#pragma unroll
    for (int e = 0; e < E; ++e) {
        float f   = fmaxf(fabsf(s[e]), m1);
        bool keep = ((m1 - s[e]) * fast_rcp(f)) <= two_eps;
        sum1 += keep ? __expf(s[e] - m1) : 0.f;
    }
    float g1 = fast_rcp(sum1);  // exp(0)/sum

    // ---- top-2 (exclude i1) + masked softmax gate
    int i2 = -1; float m2 = -3.4e38f;
#pragma unroll
    for (int e = 0; e < E; ++e) if (e != i1 && s[e] > m2) { m2 = s[e]; i2 = e; }
    float sum2 = 0.f;
#pragma unroll
    for (int e = 0; e < E; ++e) {
        if (e == i1) continue;
        float f   = fmaxf(fabsf(s[e]), m2);
        bool keep = ((m2 - s[e]) * fast_rcp(f)) <= two_eps;
        sum2 += keep ? __expf(s[e] - m2) : 0.f;
    }
    float g2 = fast_rcp(sum2);

    int slot1 = atomicAdd(&counts[i1], 1);
    toklist[i1 * T + slot1] = t;
    gatec[i1 * T + slot1]   = g1;
    int slot2 = atomicAdd(&counts[i2], 1);
    toklist[i2 * T + slot2] = t;
    gatec[i2 * T + slot2]   = g2;
}

// ---------------------------------------------------------------- kernel 2: expert FFN (WMMA bf16)
__global__ __launch_bounds__(256) void moe_expert_kernel(const float* __restrict__ x,
                                                         const float* __restrict__ wall,
                                                         const float* __restrict__ w2all,
                                                         const int* __restrict__ counts,
                                                         const int* __restrict__ toklist,
                                                         const float* __restrict__ gatec,
                                                         float* __restrict__ out) {
    const int e   = blockIdx.y;
    const int cnt = counts[e];
    const int t0  = blockIdx.x * 16;
    if (t0 >= cnt) return;  // wave-uniform exit

    extern __shared__ __align__(16) char smem[];
    bf16_t* lx     = (bf16_t*)smem;                                       // [16][LDA]
    bf16_t* linter = (bf16_t*)(smem + (size_t)16 * LDA * sizeof(bf16_t)); // [16][LDI]
    __shared__ int   ttile[16];
    __shared__ float gtile[16];

    const int tid = threadIdx.x;
    if (tid < 16) {
        int idx = t0 + tid;
        if (idx < cnt) { ttile[tid] = toklist[e * T + idx]; gtile[tid] = gatec[e * T + idx]; }
        else           { ttile[tid] = 0;                    gtile[tid] = 0.f; }
    }
    __syncthreads();

    // ---- stage gathered token rows into LDS (fp32 -> bf16)
    for (int idx = tid; idx < 16 * (H / 4); idx += 256) {
        int r  = idx >> 9;           // H/4 == 512 float4 per row
        int c4 = (idx & 511) << 2;
        v4f v = *(const v4f*)(x + (size_t)ttile[r] * H + c4);
        bf16_t* dst = lx + r * LDA + c4;
        dst[0] = (bf16_t)v[0]; dst[1] = (bf16_t)v[1];
        dst[2] = (bf16_t)v[2]; dst[3] = (bf16_t)v[3];
    }
    __syncthreads();

    const int lane = tid & 31;
    const int wv   = tid >> 5;       // wave 0..7
    const int n    = lane & 15;      // output column / A row / B row
    const int hi   = lane >> 4;
    const int c0   = hi * 8;         // A fragment k sub-chunk
    const int k0   = hi * 8;         // B fragment k sub-chunk

    const float*  wse  = wall  + (size_t)e * 2 * I * H;
    const float*  w2e  = w2all + (size_t)e * H * I;
    const bf16_t* arow = lx + n * LDA;

    // ---- GEMM1: h = x @ ws[e]^T ; fuse SiLU(gate)*up -> inter (bf16, LDS)
    for (int og = wv * 16; og < I; og += 128) {
        v8f accg = zero8();
        v8f accu = zero8();
        const float* wg = wse + (size_t)(og + n) * H;        // gate rows
        const float* wu = wse + (size_t)(I + og + n) * H;    // up rows
#pragma unroll 2
        for (int kb = 0; kb < H; kb += 32) {
            __builtin_prefetch(wg + kb + 256, 0, 0);         // global_prefetch_b8
            __builtin_prefetch(wu + kb + 256, 0, 0);
            v16bf a  = load_a_frag(arow, kb + c0);
            v16bf bg = load_b_frag(wg + kb + k0);
            v16bf bu = load_b_frag(wu + kb + k0);
            accg = wmma_bf16(a, bg, accg);
            accu = wmma_bf16_reuseA(a, bu, accu);            // identical op, same A regs
        }
#pragma unroll
        for (int r = 0; r < 8; ++r) {
            int   m = r + hi * 8;                 // C/D layout: VGPR r, halves -> M
            float g = accg[r];
            float v = g * accu[r] * fast_rcp(1.f + __expf(-g));   // SiLU(g) * up
            linter[m * LDI + og + n] = (bf16_t)v;
        }
    }
    __syncthreads();

    // ---- GEMM2: out_tile = inter @ w2s[e]^T ; scale by gate, scatter-add
    const bf16_t* irow = linter + n * LDI;
    for (int hg = wv * 16; hg < H; hg += 128) {
        v8f acc = zero8();
        const float* w2 = w2e + (size_t)(hg + n) * I;
#pragma unroll 2
        for (int kb = 0; kb < I; kb += 32) {
            __builtin_prefetch(w2 + kb + 256, 0, 0);
            v16bf a = load_a_frag(irow, kb + c0);
            v16bf b = load_b_frag(w2 + kb + k0);
            acc = wmma_bf16(a, b, acc);
        }
#pragma unroll
        for (int r = 0; r < 8; ++r) {
            int m = r + hi * 8;
            atomicAdd(out + (size_t)ttile[m] * H + hg + n, acc[r] * gtile[m]);
        }
    }
}

// ---------------------------------------------------------------- launcher
extern "C" void kernel_launch(void* const* d_in, const int* in_sizes, int n_in,
                              void* d_out, int out_size, void* d_ws, size_t ws_size,
                              hipStream_t stream) {
    (void)in_sizes; (void)n_in; (void)out_size; (void)ws_size;

    const float* x      = (const float*)d_in[0];  // (T,H)
    const float* gate_w = (const float*)d_in[1];  // (E,H)
    const float* ws     = (const float*)d_in[2];  // (E,2I,H)
    const float* w2s    = (const float*)d_in[3];  // (E,H,I)
    float* out = (float*)d_out;                   // (T,H)

    int*   counts  = (int*)d_ws;
    int*   toklist = (int*)((char*)d_ws + 256);
    float* gatec   = (float*)((char*)d_ws + 256 + (size_t)E * T * sizeof(int));

    moe_zero_kernel<<<1024, 256, 0, stream>>>(out, T * H, counts);
    moe_router_kernel<<<T / 256, 256, 0, stream>>>(x, gate_w, counts, toklist, gatec);

    dim3 grid(T / 16, E);
    moe_expert_kernel<<<grid, 256, SMEM_BYTES, stream>>>(x, ws, w2s, counts, toklist,
                                                         gatec, out);
}
